// SimpleModel_27298812133548
// MI455X (gfx1250) — compile-verified, matrix operand order checked by reading, not run on hardware
//
#include <hip/hip_runtime.h>
#include <hip/hip_bf16.h>
#include <math.h>

// ---------------------------------------------------------------------------
// Problem dims (fixed by reference)
// ---------------------------------------------------------------------------
#define B_    256
#define T_    512
#define VOC_  32000
#define E_    512
#define H_    1024
#define G4_   4096      // 4*H
#define DN_   512       // DENSE
#define NC_   64        // NUM_CLASSES

typedef __bf16 bf16;
typedef __attribute__((ext_vector_type(16))) __bf16 v16bf;
typedef __attribute__((ext_vector_type(8)))  __bf16 v8bf;
typedef __attribute__((ext_vector_type(8)))  float  v8f;
typedef __attribute__((ext_vector_type(4)))  unsigned int u32x4;
typedef __attribute__((ext_vector_type(4)))  int i32x4;
typedef __attribute__((ext_vector_type(8)))  int i32x8;
typedef __attribute__((ext_vector_type(4)))  unsigned int uint4v;

#if defined(__has_builtin)
#if __has_builtin(__builtin_amdgcn_tensor_load_to_lds)
#define HAVE_TDM 1
#else
#define HAVE_TDM 0
#endif
#else
#define HAVE_TDM 0
#endif

// ---------------------------------------------------------------------------
// WMMA wrapper: D = A(16x32 bf16) x B(32x16 bf16) + C(16x16 f32)
// ---------------------------------------------------------------------------
__device__ __forceinline__ v8f wmma_bf16(v16bf a, v16bf b, v8f c) {
  return __builtin_amdgcn_wmma_f32_16x16x32_bf16(
      /*neg_a=*/false, a, /*neg_b=*/false, b,
      /*c_mod=*/(short)0, c, /*reuse_a=*/false, /*reuse_b=*/false);
}

// 16-bit A-matrix 16x32 lane/half mapping (ISA 7.12.2):
//   lanes 0-15  : halves 0..7 -> K=0..7,   halves 8..15 -> K=16..23
//   lanes 16-31 : halves 0..7 -> K=8..15,  halves 8..15 -> K=24..31
__device__ __forceinline__ int klocal(int lane, int j) {
  return ((j >= 8) ? 16 : 0) + (j & 7) + ((lane >= 16) ? 8 : 0);
}

// ---------------------------------------------------------------------------
// TDM: 1D contiguous copy of `ndw` dwords from global to LDS (one wave issues).
// D# built per ISA ch.8: group0 = {count=1 | lds_addr | global_addr | type=2},
// group1 = {data_size=4B, tensor_dim0=ndw, tensor_dim1=1, tile_dim0=ndw,
//           tensor_dim0_stride=ndw}.  Groups 2/3 unused (2D-capable form).
// ---------------------------------------------------------------------------
__device__ __forceinline__ void tdm_load_1d(unsigned lds_off, const void* src,
                                            unsigned ndw) {
#if HAVE_TDM
  unsigned long long ga = (unsigned long long)(size_t)src;
  u32x4 g0;
  g0[0] = 1u;                                   // count=1 (user descriptor)
  g0[1] = lds_off;                              // lds_addr (bytes)
  g0[2] = (unsigned)(ga & 0xffffffffu);         // global_addr[31:0]
  g0[3] = (unsigned)((ga >> 32) & 0x1ffffffu)   // global_addr[56:32]
          | (2u << 30);                         // type=2 ("image")
  i32x8 g1;
  g1[0] = (int)(2u << 16);                      // data_size=4B, wg_mask=0
  g1[1] = (int)(ndw << 16);                     // tensor_dim0[15:0]
  g1[2] = (int)((ndw >> 16) | (1u << 16));      // tensor_dim0[31:16], dim1=1
  g1[3] = (int)(ndw << 16);                     // tile_dim0 = ndw
  g1[4] = 0;                                    // tile_dim1/2 unused
  g1[5] = (int)ndw;                             // tensor_dim0_stride lo
  g1[6] = 0;
  g1[7] = 0;
  i32x4 gz = {0, 0, 0, 0};
#if __clang_major__ >= 23
  i32x8 gz8 = {0, 0, 0, 0, 0, 0, 0, 0};
  __builtin_amdgcn_tensor_load_to_lds(g0, g1, gz, gz, gz8, 0);
#else
  __builtin_amdgcn_tensor_load_to_lds(g0, g1, gz, gz, 0);
#endif
#endif
}

__device__ __forceinline__ void wait_tensorcnt0() {
#if defined(__has_builtin) && __has_builtin(__builtin_amdgcn_s_wait_tensorcnt)
  __builtin_amdgcn_s_wait_tensorcnt(0);
#else
  asm volatile("s_wait_tensorcnt 0x0" ::: "memory");
#endif
}

// Stage `bytes` (multiple of 16, <=256KB) of A-fragments into LDS.
// TDM path: wave 0 issues one DMA; fallback: cooperative vector copy.
__device__ __forceinline__ void stage_to_lds(bf16* sh, const bf16* gsrc,
                                             int bytes) {
#if HAVE_TDM
  if ((threadIdx.x >> 5) == 0) {
    tdm_load_1d((unsigned)(size_t)(void*)sh, (const void*)gsrc,
                (unsigned)(bytes >> 2));
    wait_tensorcnt0();
  }
#else
  const uint4v* s = (const uint4v*)gsrc;
  uint4v* d = (uint4v*)sh;
  int n = bytes >> 4;
  for (int i = threadIdx.x; i < n; i += blockDim.x) d[i] = s[i];
#endif
  __syncthreads();
}

// ---------------------------------------------------------------------------
// Pack a row-major f32 weight matrix (K x N) into bf16 B-fragments.
// Fragment (nt, kt) id = nt*(K/32)+kt ; per-lane 16 contiguous halves.
// ---------------------------------------------------------------------------
__global__ void pack_wB(const float* __restrict__ src, bf16* __restrict__ dst,
                        int K, int N) {
  int idx  = blockIdx.x * blockDim.x + threadIdx.x;
  int KT   = K >> 5;
  int NT   = N >> 4;
  int lane = idx & 31;
  int frag = idx >> 5;
  if (frag >= NT * KT) return;
  int nt = frag / KT, kt = frag % KT;
  int n  = nt * 16 + (lane & 15);
  v16bf v;
#pragma unroll
  for (int j = 0; j < 16; ++j) {
    int k = kt * 32 + klocal(lane, j);
    v[j] = (bf16)src[(size_t)k * N + n];
  }
  *(v16bf*)(dst + (size_t)idx * 16) = v;
}

// ---------------------------------------------------------------------------
// Embedding gather + pack X (rows ordered r = t*B + b) into bf16 A-fragments.
// Fragment (mt, kt) id = mt*16+kt (KT = E/32 = 16).
// ---------------------------------------------------------------------------
__global__ void gather_pack_x(const long long* __restrict__ tokens,
                              const float* __restrict__ emb,
                              bf16* __restrict__ xp) {
  int idx  = blockIdx.x * blockDim.x + threadIdx.x;
  int lane = idx & 31;
  int frag = idx >> 5;          // 0 .. 8192*16-1
  int mt   = frag >> 4;
  int kt   = frag & 15;
  int row  = mt * 16 + (lane & 15);   // row = t*B + b
  int t    = row >> 8;
  int b    = row & 255;
  long long tok = tokens[(size_t)b * T_ + t];
  const float* er = emb + (size_t)tok * E_;
  v16bf v;
#pragma unroll
  for (int j = 0; j < 16; ++j) v[j] = (bf16)er[kt * 32 + klocal(lane, j)];
  *(v16bf*)(xp + (size_t)idx * 16) = v;
}

// ---------------------------------------------------------------------------
// maskT[t*B+b] = tokens[b,t] != 0
// ---------------------------------------------------------------------------
__global__ void mask_init(const long long* __restrict__ tokens,
                          float* __restrict__ maskT) {
  int idx = blockIdx.x * blockDim.x + threadIdx.x;
  if (idx >= B_ * T_) return;
  int t = idx >> 8, b = idx & 255;
  maskT[idx] = (tokens[(size_t)b * T_ + t] != 0) ? 1.0f : 0.0f;
}

__global__ void zero_state(float* __restrict__ cp, bf16* __restrict__ hA0) {
  int idx = blockIdx.x * blockDim.x + threadIdx.x;
  if (idx >= B_ * H_) return;
  cp[idx]  = 0.0f;
  hA0[idx] = (bf16)0.0f;
}

// ---------------------------------------------------------------------------
// ZX = X @ W  (131072 x 4096), bf16 in / f32 acc / bf16 out.
// One wave computes a 16x64 strip (4 N-tiles). A tile (16KB) TDM-staged in
// LDS and shared by the block's 8 waves; B double-buffered in registers.
// ---------------------------------------------------------------------------
__global__ void __launch_bounds__(256)
gemm_xw(const bf16* __restrict__ xp, const bf16* __restrict__ wp,
        bf16* __restrict__ zx) {
  __shared__ bf16 shA[8192];          // 16 frags x 512 halves = 16 KB
  int wv   = threadIdx.x >> 5;
  int lane = threadIdx.x & 31;
  int mt   = (blockIdx.x * 8) >> 6;   // same for all 8 waves of the block
  int ntg  = (blockIdx.x * 8 + wv) & 63;
  const int KT = 16;                  // E/32

  stage_to_lds(shA, xp + (size_t)(mt * KT) * 512, KT * 512 * 2);

  v8f z = {0.f,0.f,0.f,0.f,0.f,0.f,0.f,0.f};
  v8f acc[4] = {z, z, z, z};
  v16bf b0[4], b1[4];
#pragma unroll
  for (int q = 0; q < 4; ++q)
    b0[q] = *(const v16bf*)(wp + (size_t)(((ntg * 4 + q) * KT) * 32 + lane) * 16);
  for (int kt = 0; kt < KT; kt += 2) {
#pragma unroll
    for (int q = 0; q < 4; ++q)
      b1[q] = *(const v16bf*)(wp +
          (size_t)(((ntg * 4 + q) * KT + kt + 1) * 32 + lane) * 16);
    v16bf a0 = *(const v16bf*)(shA + (size_t)(kt * 32 + lane) * 16);
#pragma unroll
    for (int q = 0; q < 4; ++q) acc[q] = wmma_bf16(a0, b0[q], acc[q]);
    if (kt + 2 < KT) {
#pragma unroll
      for (int q = 0; q < 4; ++q)
        b0[q] = *(const v16bf*)(wp +
            (size_t)(((ntg * 4 + q) * KT + kt + 2) * 32 + lane) * 16);
    }
    v16bf a1 = *(const v16bf*)(shA + (size_t)((kt + 1) * 32 + lane) * 16);
#pragma unroll
    for (int q = 0; q < 4; ++q) acc[q] = wmma_bf16(a1, b1[q], acc[q]);
  }
#pragma unroll
  for (int q = 0; q < 4; ++q) {
    v8bf o;
#pragma unroll
    for (int r = 0; r < 8; ++r) o[r] = (bf16)acc[q][r];
    *(v8bf*)(zx + ((size_t)(mt * 256 + ntg * 4 + q) * 32 + lane) * 8) = o;
  }
}

// ---------------------------------------------------------------------------
// Fused recurrent step: z = h@U (+ precomputed x@W + b), gates, masked state
// update. One wave owns a 16x16 h/c tile and all 4 gate accumulations.
// h A-tile (32KB) TDM-staged in LDS; U fragments double-buffered in registers
// (this kernel runs at ~1-2 waves/SIMD, so intra-wave pipelining matters).
// hA double buffered across steps (src -> dst).
// ---------------------------------------------------------------------------
__global__ void __launch_bounds__(256)
lstm_step(const bf16* __restrict__ hA_src, bf16* __restrict__ hA_dst,
          float* __restrict__ cp, const bf16* __restrict__ Up,
          const bf16* __restrict__ zx, const float* __restrict__ bias,
          const float* __restrict__ maskT, int t) {
  __shared__ bf16 shA[16384];          // 32 frags x 512 halves = 32 KB
  int wv   = threadIdx.x >> 5;
  int lane = threadIdx.x & 31;
  int mt   = blockIdx.y;               // 0..15  (B tiles), shared by block
  int nt   = blockIdx.x * 8 + wv;      // 0..63  (H tiles)
  const int KT = 32;                   // H/32

  stage_to_lds(shA, hA_src + (size_t)(mt * KT) * 512, KT * 512 * 2);

  v8f z = {0.f,0.f,0.f,0.f,0.f,0.f,0.f,0.f};
  v8f acc[4] = {z, z, z, z};
  v16bf b0[4], b1[4];
#pragma unroll
  for (int g = 0; g < 4; ++g)
    b0[g] = *(const v16bf*)(Up + (size_t)(((g * 64 + nt) * KT) * 32 + lane) * 16);
  for (int kt = 0; kt < KT; kt += 2) {
#pragma unroll
    for (int g = 0; g < 4; ++g)
      b1[g] = *(const v16bf*)(Up +
          (size_t)(((g * 64 + nt) * KT + kt + 1) * 32 + lane) * 16);
    v16bf a0 = *(const v16bf*)(shA + (size_t)(kt * 32 + lane) * 16);
#pragma unroll
    for (int g = 0; g < 4; ++g) acc[g] = wmma_bf16(a0, b0[g], acc[g]);
    if (kt + 2 < KT) {
#pragma unroll
      for (int g = 0; g < 4; ++g)
        b0[g] = *(const v16bf*)(Up +
            (size_t)(((g * 64 + nt) * KT + kt + 2) * 32 + lane) * 16);
    }
    v16bf a1 = *(const v16bf*)(shA + (size_t)((kt + 1) * 32 + lane) * 16);
#pragma unroll
    for (int g = 0; g < 4; ++g) acc[g] = wmma_bf16(a1, b1[g], acc[g]);
  }

  // ---- epilogue ----
  int lhalf = (lane >= 16) ? 8 : 0;
  int lc    = lane & 15;
  int n     = nt * 16 + lc;
  float bi = bias[0 * H_ + n], bff = bias[1 * H_ + n];
  float bg = bias[2 * H_ + n], bo  = bias[3 * H_ + n];
  int mtg = t * 16 + mt;
  v8bf zxi = *(const v8bf*)(zx + ((size_t)(mtg * 256 +   0 + nt) * 32 + lane) * 8);
  v8bf zxf = *(const v8bf*)(zx + ((size_t)(mtg * 256 +  64 + nt) * 32 + lane) * 8);
  v8bf zxg = *(const v8bf*)(zx + ((size_t)(mtg * 256 + 128 + nt) * 32 + lane) * 8);
  v8bf zxo = *(const v8bf*)(zx + ((size_t)(mtg * 256 + 192 + nt) * 32 + lane) * 8);
  float* cfr = cp + ((size_t)(mt * 64 + nt) * 32 + lane) * 8;
  // scatter coordinates of this lane's h elements inside the A-fragment buffer
  int ktd  = nt >> 1;
  int jo   = ((nt & 1) ? 8 : 0) + (lc & 7);
  int dlhi = (lc >= 8) ? 16 : 0;
#pragma unroll
  for (int r = 0; r < 8; ++r) {
    int mloc = r + lhalf;   // 0..15 row inside tile
    size_t hoff = ((size_t)((mt * KT + ktd) * 32) + (mloc + dlhi)) * 16 + jo;
    float ho = (float)hA_src[hoff];
    float co = cfr[r];
    float zi = acc[0][r] + (float)zxi[r] + bi;
    float zf = acc[1][r] + (float)zxf[r] + bff;
    float zg = acc[2][r] + (float)zxg[r] + bg;
    float zo = acc[3][r] + (float)zxo[r] + bo;
    float ig = 1.0f / (1.0f + __expf(-zi));
    float fg = 1.0f / (1.0f + __expf(-zf));
    float gg = tanhf(zg);
    float og = 1.0f / (1.0f + __expf(-zo));
    float cn = fg * co + ig * gg;
    float hn = og * tanhf(cn);
    float m  = maskT[t * B_ + mt * 16 + mloc];
    float cs = m * cn + (1.0f - m) * co;
    float hs = m * hn + (1.0f - m) * ho;
    cfr[r]       = cs;
    hA_dst[hoff] = (bf16)hs;
  }
}

// ---------------------------------------------------------------------------
// Repack final c (packed f32 C-fragments) into bf16 A-fragments for the head.
// ---------------------------------------------------------------------------
__global__ void repack_cA(const float* __restrict__ cp, bf16* __restrict__ cA) {
  int idx  = blockIdx.x * blockDim.x + threadIdx.x;  // frag*32+lane
  int lane = idx & 31;
  int frag = idx >> 5;       // 0..511 (16 M-tiles x 32 K-tiles)
  if (frag >= 512) return;
  int mt = frag >> 5;
  int kt = frag & 31;
  int m  = mt * 16 + (lane & 15);
  v16bf v;
#pragma unroll
  for (int j = 0; j < 16; ++j) {
    int k     = kt * 32 + klocal(lane, j);        // column in H
    int ntc   = k >> 4;
    int lanec = (k & 15) + (((m & 15) >= 8) ? 16 : 0);
    int rr    = (m & 15) & 7;
    v[j] = (bf16)cp[((size_t)(mt * 64 + ntc) * 32 + lanec) * 8 + rr];
  }
  *(v16bf*)(cA + (size_t)idx * 16) = v;
}

// ---------------------------------------------------------------------------
// hid = relu([h,c] @ W1 + b1)  -> stored directly as bf16 A-fragments.
// M=256 N=512 K=2048.  A-frags: kt<32 from hA, kt>=32 from cA.
// ---------------------------------------------------------------------------
__global__ void __launch_bounds__(256)
classifier1(const bf16* __restrict__ hA, const bf16* __restrict__ cA,
            const bf16* __restrict__ w1p, const float* __restrict__ b1,
            bf16* __restrict__ hidA) {
  int wgid = blockIdx.x * 8 + (threadIdx.x >> 5);  // 0..511
  int lane = threadIdx.x & 31;
  int mt   = wgid >> 5;   // 0..15
  int nt   = wgid & 31;   // 0..31
  const int KT = 64;      // 2048/32
  v8f acc = {0.f,0.f,0.f,0.f,0.f,0.f,0.f,0.f};
  for (int kt = 0; kt < KT; ++kt) {
    const bf16* ap = (kt < 32)
        ? (hA + (size_t)((mt * 32 + kt)        * 32 + lane) * 16)
        : (cA + (size_t)((mt * 32 + (kt - 32)) * 32 + lane) * 16);
    v16bf a  = *(const v16bf*)ap;
    v16bf bb = *(const v16bf*)(w1p + (size_t)((nt * KT + kt) * 32 + lane) * 16);
    acc = wmma_bf16(a, bb, acc);
  }
  int lc = lane & 15, lh = (lane >= 16) ? 8 : 0;
  float bias = b1[nt * 16 + lc];
  int ktd  = nt >> 1;
  int jo   = ((nt & 1) ? 8 : 0) + (lc & 7);
  int dlhi = (lc >= 8) ? 16 : 0;
#pragma unroll
  for (int r = 0; r < 8; ++r) {
    float v = acc[r] + bias;
    v = v > 0.0f ? v : 0.0f;
    int mloc = r + lh;
    hidA[((size_t)((mt * 16 + ktd) * 32) + (mloc + dlhi)) * 16 + jo] = (bf16)v;
  }
}

// ---------------------------------------------------------------------------
// logits = hid @ W2 + b2.  M=256 N=64 K=512.  Row-major f32 output.
// ---------------------------------------------------------------------------
__global__ void __launch_bounds__(256)
classifier2(const bf16* __restrict__ hidA, const bf16* __restrict__ w2p,
            const float* __restrict__ b2, float* __restrict__ logits) {
  int wgid = blockIdx.x * 8 + (threadIdx.x >> 5);  // 0..63
  int lane = threadIdx.x & 31;
  int mt   = wgid >> 2;   // 0..15
  int nt   = wgid & 3;    // 0..3
  const int KT = 16;      // 512/32
  v8f acc = {0.f,0.f,0.f,0.f,0.f,0.f,0.f,0.f};
  for (int kt = 0; kt < KT; ++kt) {
    v16bf a  = *(const v16bf*)(hidA + (size_t)((mt * KT + kt) * 32 + lane) * 16);
    v16bf bb = *(const v16bf*)(w2p  + (size_t)((nt * KT + kt) * 32 + lane) * 16);
    acc = wmma_bf16(a, bb, acc);
  }
  int lc = lane & 15, lh = (lane >= 16) ? 8 : 0;
  int n = nt * 16 + lc;
  float bias = b2[n];
#pragma unroll
  for (int r = 0; r < 8; ++r) {
    int row = mt * 16 + r + lh;
    logits[(size_t)row * NC_ + n] = acc[r] + bias;
  }
}

// ---------------------------------------------------------------------------
// Row softmax over 64 classes. One 64-thread block per row.
// ---------------------------------------------------------------------------
__global__ void softmax64(const float* __restrict__ logits,
                          float* __restrict__ out) {
  __shared__ float s[64];
  int row = blockIdx.x, c = threadIdx.x;
  float v = logits[(size_t)row * NC_ + c];
  s[c] = v;
  __syncthreads();
  for (int off = 32; off > 0; off >>= 1) {
    if (c < off) s[c] = fmaxf(s[c], s[c + off]);
    __syncthreads();
  }
  float mx = s[0];
  __syncthreads();
  float e = __expf(v - mx);
  s[c] = e;
  __syncthreads();
  for (int off = 32; off > 0; off >>= 1) {
    if (c < off) s[c] = s[c] + s[c + off];
    __syncthreads();
  }
  out[(size_t)row * NC_ + c] = e / s[0];
}

// ---------------------------------------------------------------------------
// Host driver
// ---------------------------------------------------------------------------
extern "C" void kernel_launch(void* const* d_in, const int* in_sizes, int n_in,
                              void* d_out, int out_size, void* d_ws, size_t ws_size,
                              hipStream_t stream) {
  const long long* tokens = (const long long*)d_in[0];
  const float* emb = (const float*)d_in[1];
  const float* W   = (const float*)d_in[2];
  const float* U   = (const float*)d_in[3];
  const float* b   = (const float*)d_in[4];
  const float* W1  = (const float*)d_in[5];
  const float* b1  = (const float*)d_in[6];
  const float* W2  = (const float*)d_in[7];
  const float* b2  = (const float*)d_in[8];
  float* out = (float*)d_out;

  // workspace carve-up (256B aligned)
  size_t off = 0;
  auto carve = [&](size_t bytes) -> char* {
    char* p = (char*)d_ws + off;
    off += (bytes + 255) & ~(size_t)255;
    return p;
  };
  bf16*  Xp    = (bf16*) carve((size_t)B_ * T_ * E_ * 2);     // 134 MB
  bf16*  Wp    = (bf16*) carve((size_t)E_ * G4_ * 2);         //   4 MB
  bf16*  Up    = (bf16*) carve((size_t)H_ * G4_ * 2);         //   8 MB
  bf16*  W1p   = (bf16*) carve((size_t)(2 * H_) * DN_ * 2);   //   2 MB
  bf16*  W2p   = (bf16*) carve((size_t)DN_ * NC_ * 2);        //  64 KB
  bf16*  ZX    = (bf16*) carve((size_t)B_ * T_ * G4_ * 2);    // 1.07 GB
  float* maskT = (float*)carve((size_t)B_ * T_ * 4);
  float* cp    = (float*)carve((size_t)B_ * H_ * 4);
  bf16*  hA0   = (bf16*) carve((size_t)B_ * H_ * 2);
  bf16*  hA1   = (bf16*) carve((size_t)B_ * H_ * 2);
  bf16*  cA    = (bf16*) carve((size_t)B_ * H_ * 2);
  bf16*  hidA  = (bf16*) carve((size_t)B_ * DN_ * 2);
  float* logit = (float*)carve((size_t)B_ * NC_ * 4);

  // 1) pack weights into WMMA B-fragments (bf16)
  pack_wB<<<(16 * 256 * 32) / 256, 256, 0, stream>>>(W,  Wp,  E_,     G4_);
  pack_wB<<<(32 * 256 * 32) / 256, 256, 0, stream>>>(U,  Up,  H_,     G4_);
  pack_wB<<<(64 *  32 * 32) / 256, 256, 0, stream>>>(W1, W1p, 2 * H_, DN_);
  pack_wB<<<(16 *   4 * 32) / 256, 256, 0, stream>>>(W2, W2p, DN_,    NC_);

  // 2) mask + embedding gather/pack + zero state
  mask_init<<<(B_ * T_) / 256, 256, 0, stream>>>(tokens, maskT);
  gather_pack_x<<<(8192 * 16 * 32) / 256, 256, 0, stream>>>(tokens, emb, Xp);
  zero_state<<<(B_ * H_) / 256, 256, 0, stream>>>(cp, hA0);

  // 3) big parallel GEMM: ZX = X @ W  (all timesteps)
  gemm_xw<<<(8192 * 64) / 8, 256, 0, stream>>>(Xp, Wp, ZX);

  // 4) recurrence: 512 fused steps, double-buffered packed h
  for (int t = 0; t < T_; ++t) {
    bf16* s = (t & 1) ? hA1 : hA0;
    bf16* d = (t & 1) ? hA0 : hA1;
    lstm_step<<<dim3(8, 16), 256, 0, stream>>>(s, d, cp, Up, ZX, b, maskT, t);
  }
  // after t=511, final h is in hA0

  // 5) classifier head
  repack_cA<<<(512 * 32) / 256, 256, 0, stream>>>(cp, cA);
  classifier1<<<(16 * 32) * 32 / 256, 256, 0, stream>>>(hA0, cA, W1p, b1, hidA);
  classifier2<<<(16 * 4) * 32 / 256, 256, 0, stream>>>(hidA, W2p, b2, logit);
  softmax64<<<B_, 64, 0, stream>>>(logit, out);
}
